// RecurrentBase_23218593202568
// MI455X (gfx1250) — matrix operand order repaired and checked by reading, not compiled, not run on hardware
//
#include <hip/hip_runtime.h>
#include <hip/hip_bf16.h>

#define T_STEPS 512
#define BATCH   64
#define IN      512
#define HID     1024
#define G4      4096
#define K0      (IN + HID)   // 1536
#define K1      (2 * HID)    // 2048
#define NWG     64
#define NTHR    256
#define HS      (HID / NWG)  // 16 h-columns per workgroup

typedef __bf16 bf16_t;
typedef __bf16 v16bf __attribute__((ext_vector_type(16)));
typedef float  v8f   __attribute__((ext_vector_type(8)));
typedef unsigned int u32x4 __attribute__((ext_vector_type(4)));

union Frag { u32x4 u[2]; v16bf v; };

// ---------------- prep kernels ----------------

__global__ void k_cvt_x(const float* __restrict__ x, bf16_t* __restrict__ xb, int n) {
  for (int i = blockIdx.x * blockDim.x + threadIdx.x; i < n; i += gridDim.x * blockDim.x)
    xb[i] = (bf16_t)x[i];
}

// W0[n][k] = k<IN ? W_ih0[n][k] : W_hh0[n][k-IN]   (rows = gate rows, K contiguous)
// W1[n][k] = k<HID? W_ih1[n][k] : W_hh1[n][k-HID]
__global__ void k_pack_w(const float* __restrict__ Wih0, const float* __restrict__ Whh0,
                         const float* __restrict__ Wih1, const float* __restrict__ Whh1,
                         bf16_t* __restrict__ W0, bf16_t* __restrict__ W1) {
  const int tot0 = G4 * K0, tot1 = G4 * K1;
  for (int i = blockIdx.x * blockDim.x + threadIdx.x; i < tot0 + tot1;
       i += gridDim.x * blockDim.x) {
    if (i < tot0) {
      int n = i / K0, k = i % K0;
      W0[i] = (bf16_t)(k < IN ? Wih0[n * IN + k] : Whh0[n * HID + (k - IN)]);
    } else {
      int j = i - tot0;
      int n = j / K1, k = j % K1;
      W1[j] = (bf16_t)(k < HID ? Wih1[n * HID + k] : Whh1[n * HID + (k - HID)]);
    }
  }
}

__global__ void k_init(const float* __restrict__ bih0, const float* __restrict__ bhh0,
                       const float* __restrict__ bih1, const float* __restrict__ bhh1,
                       float* __restrict__ b0, float* __restrict__ b1,
                       bf16_t* __restrict__ h0a, bf16_t* __restrict__ h0b,
                       bf16_t* __restrict__ h1a, bf16_t* __restrict__ h1b,
                       float* __restrict__ c0, float* __restrict__ c1,
                       unsigned* __restrict__ cnt) {
  const int n = BATCH * HID;  // 65536
  for (int i = blockIdx.x * blockDim.x + threadIdx.x; i < n; i += gridDim.x * blockDim.x) {
    if (i < G4) { b0[i] = bih0[i] + bhh0[i]; b1[i] = bih1[i] + bhh1[i]; }
    h0a[i] = (bf16_t)0.0f; h0b[i] = (bf16_t)0.0f;
    h1a[i] = (bf16_t)0.0f; h1b[i] = (bf16_t)0.0f;
    c0[i] = 0.0f; c1[i] = 0.0f;
    if (i == 0) *cnt = 0u;
  }
}

// ---------------- persistent LSTM kernel ----------------

__device__ __forceinline__ float sigm(float x) { return 1.0f / (1.0f + __expf(-x)); }

__device__ __forceinline__ void grid_barrier(unsigned* cnt, unsigned target) {
  __syncthreads();
  if (threadIdx.x == 0) {
    __threadfence();
    __hip_atomic_fetch_add(cnt, 1u, __ATOMIC_RELEASE, __HIP_MEMORY_SCOPE_AGENT);
    while (__hip_atomic_load(cnt, __ATOMIC_ACQUIRE, __HIP_MEMORY_SCOPE_AGENT) < target)
      __builtin_amdgcn_s_sleep(2);
  }
  __syncthreads();
}

// One layer: this wave computes a 32(M-half) x 16 gate strip over full K,
// then the WG fuses i/f/g/o for its 16-column h-slice.
__device__ __forceinline__ void layer_body(
    const bf16_t* __restrict__ As, int K,
    const bf16_t* __restrict__ W, const float* __restrict__ bias,
    float* __restrict__ cbuf, bf16_t* __restrict__ hnew, float* __restrict__ outp,
    float* __restrict__ Gs, int tid, int lane, int grp, int mh, int col0) {
  const int n16 = lane & 15;
  const int hi  = lane >> 4;
  const int selA = hi * 8;    // A frag: chunks at kt+selA, kt+16+selA (ISA 16-bit A layout)
  const int selB = hi * 8;    // B frag: chunks at kt+selB, kt+16+selB (same K order as A)

  const int grow = grp * HID + col0 + n16;   // gate row == B column for this lane
  const bf16_t* wrow = W + (size_t)grow * K;

  v8f acc[2];
#pragma unroll
  for (int q = 0; q < 2; ++q) {
    v8f z = {0.f,0.f,0.f,0.f,0.f,0.f,0.f,0.f};
    acc[q] = z;
  }

  for (int kt = 0; kt < K; kt += 32) {
    Frag bfr;
    bfr.u[0] = *(const u32x4*)&wrow[kt + selB];
    bfr.u[1] = *(const u32x4*)&wrow[kt + selB + 16];
    __builtin_prefetch(&wrow[kt + 256], 0, 1);   // weights are L2-hot
#pragma unroll
    for (int q = 0; q < 2; ++q) {
      const int mt = mh * 2 + q;
      Frag afr;
      const bf16_t* arow = &As[(size_t)(mt * 16 + n16) * K + kt];
      afr.u[0] = *(const u32x4*)&arow[selA];
      afr.u[1] = *(const u32x4*)&arow[16 + selA];
      acc[q] = __builtin_amdgcn_wmma_f32_16x16x32_bf16(
          false, afr.v, false, bfr.v, (short)0, acc[q], false, false);
    }
  }

  // Gates to LDS: Gs[group][m][HS]; C/D layout: VGPR e -> M = e + 8*hi, N = lane&15
#pragma unroll
  for (int q = 0; q < 2; ++q) {
    const int mt = mh * 2 + q;
#pragma unroll
    for (int e = 0; e < 8; ++e)
      Gs[(grp * BATCH + mt * 16 + e + 8 * hi) * HS + n16] = acc[q][e];
  }
  __syncthreads();

  // Fused i/f/g/o gating for this WG's [64 x 16] slice
  for (int idx = tid; idx < BATCH * HS; idx += NTHR) {
    int m = idx / HS, c = idx % HS, col = col0 + c;
    float ig = Gs[(0 * BATCH + m) * HS + c] + bias[col];
    float fg = Gs[(1 * BATCH + m) * HS + c] + bias[HID + col];
    float gg = Gs[(2 * BATCH + m) * HS + c] + bias[2 * HID + col];
    float og = Gs[(3 * BATCH + m) * HS + c] + bias[3 * HID + col];
    float cold = cbuf[m * HID + col];
    float cn = sigm(fg) * cold + sigm(ig) * tanhf(gg);
    float hn = sigm(og) * tanhf(cn);
    cbuf[m * HID + col] = cn;
    hnew[m * HID + col] = (bf16_t)hn;
    if (outp) outp[m * HID + col] = hn;
  }
  __syncthreads();  // Gs/As reused next phase
}

__global__ void __launch_bounds__(NTHR, 1)
k_lstm(const bf16_t* __restrict__ xb,
       const bf16_t* __restrict__ W0, const bf16_t* __restrict__ W1,
       const float* __restrict__ b0, const float* __restrict__ b1,
       bf16_t* h0a, bf16_t* h0b, bf16_t* h1a, bf16_t* h1b,
       float* c0, float* c1,
       unsigned* cnt, float* __restrict__ out) {
  extern __shared__ __align__(16) char smem[];
  bf16_t* As = (bf16_t*)smem;                                        // 64 x K panel (<=256KB)
  float*  Gs = (float*)(smem + (size_t)BATCH * K1 * sizeof(bf16_t)); // 4x64x16 gates (16KB)

  const int tid  = threadIdx.x;
  const int lane = tid & 31;
  const int wv   = tid >> 5;   // 8 waves
  const int grp  = wv >> 1;    // gate group (i,f,g,o)
  const int mh   = wv & 1;     // M half (batch rows 0-31 / 32-63)
  const int col0 = blockIdx.x * HS;

  bf16_t* h0p[2] = {h0a, h0b};
  bf16_t* h1p[2] = {h1a, h1b};

  unsigned target = 0;

  for (int t = 0; t < T_STEPS; ++t) {
    const int pn = t & 1;        // parity of buffers written this step
    const int po = pn ^ 1;       // parity read this step
    bf16_t* h0_old = h0p[po];  bf16_t* h0_new = h0p[pn];
    bf16_t* h1_old = h1p[po];  bf16_t* h1_new = h1p[pn];

    // ------- layer 0: A = [x_t | h0(t-1)], K = 1536 -------
    {
      const bf16_t* xsrc = xb + (size_t)t * BATCH * IN;
      for (int i = tid; i < BATCH * IN / 8; i += NTHR) {
        int m = i / (IN / 8), kc = i % (IN / 8);
        *(u32x4*)&As[m * K0 + kc * 8] = *(const u32x4*)&xsrc[m * IN + kc * 8];
      }
      for (int i = tid; i < BATCH * HID / 8; i += NTHR) {
        int m = i / (HID / 8), kc = i % (HID / 8);
        *(u32x4*)&As[m * K0 + IN + kc * 8] = *(const u32x4*)&h0_old[m * HID + kc * 8];
      }
    }
    __syncthreads();   // ping-pong: no grid barrier needed between stage and compute

    layer_body(As, K0, W0, b0, c0, h0_new, nullptr, Gs, tid, lane, grp, mh, col0);
    target += NWG; grid_barrier(cnt, target);   // h0(t) published

    // ------- layer 1: A = [h0(t) | h1(t-1)], K = 2048 -------
    {
      for (int i = tid; i < BATCH * HID / 8; i += NTHR) {
        int m = i / (HID / 8), kc = i % (HID / 8);
        *(u32x4*)&As[m * K1 + kc * 8]       = *(const u32x4*)&h0_new[m * HID + kc * 8];
        *(u32x4*)&As[m * K1 + HID + kc * 8] = *(const u32x4*)&h1_old[m * HID + kc * 8];
      }
    }
    __syncthreads();

    layer_body(As, K1, W1, b1, c1, h1_new, out + (size_t)t * BATCH * HID, Gs,
               tid, lane, grp, mh, col0);
    target += NWG; grid_barrier(cnt, target);   // h1(t) published
  }
}

// ---------------- host ----------------

extern "C" void kernel_launch(void* const* d_in, const int* in_sizes, int n_in,
                              void* d_out, int out_size, void* d_ws, size_t ws_size,
                              hipStream_t stream) {
  const float* x    = (const float*)d_in[0];
  const float* Wih0 = (const float*)d_in[1];
  const float* Whh0 = (const float*)d_in[2];
  const float* bih0 = (const float*)d_in[3];
  const float* bhh0 = (const float*)d_in[4];
  const float* Wih1 = (const float*)d_in[5];
  const float* Whh1 = (const float*)d_in[6];
  const float* bih1 = (const float*)d_in[7];
  const float* bhh1 = (const float*)d_in[8];
  float* out = (float*)d_out;

  char* ws = (char*)d_ws;
  size_t off = 0;
  auto take = [&](size_t bytes) { char* p = ws + off; off = (off + bytes + 255) & ~(size_t)255; return p; };

  bf16_t* xb  = (bf16_t*)take((size_t)T_STEPS * BATCH * IN * sizeof(bf16_t)); // 32 MB
  bf16_t* W0  = (bf16_t*)take((size_t)G4 * K0 * sizeof(bf16_t));              // 12 MB
  bf16_t* W1  = (bf16_t*)take((size_t)G4 * K1 * sizeof(bf16_t));              // 16 MB
  float*  b0  = (float*)take((size_t)G4 * sizeof(float));
  float*  b1  = (float*)take((size_t)G4 * sizeof(float));
  bf16_t* h0a = (bf16_t*)take((size_t)BATCH * HID * sizeof(bf16_t));
  bf16_t* h0b = (bf16_t*)take((size_t)BATCH * HID * sizeof(bf16_t));
  bf16_t* h1a = (bf16_t*)take((size_t)BATCH * HID * sizeof(bf16_t));
  bf16_t* h1b = (bf16_t*)take((size_t)BATCH * HID * sizeof(bf16_t));
  float*  c0  = (float*)take((size_t)BATCH * HID * sizeof(float));
  float*  c1  = (float*)take((size_t)BATCH * HID * sizeof(float));
  unsigned* cnt = (unsigned*)take(sizeof(unsigned));
  (void)ws_size; (void)in_sizes; (void)n_in; (void)out_size;

  k_cvt_x<<<4096, 256, 0, stream>>>(x, xb, T_STEPS * BATCH * IN);
  k_pack_w<<<4096, 256, 0, stream>>>(Wih0, Whh0, Wih1, Whh1, W0, W1);
  k_init<<<256, 256, 0, stream>>>(bih0, bhh0, bih1, bhh1, b0, b1,
                                  h0a, h0b, h1a, h1b, c0, c1, cnt);

  const int smem_bytes = BATCH * K1 * (int)sizeof(bf16_t) + 4 * BATCH * HS * (int)sizeof(float); // 278528
  hipFuncSetAttribute((const void*)k_lstm, hipFuncAttributeMaxDynamicSharedMemorySize, smem_bytes);
  k_lstm<<<NWG, NTHR, smem_bytes, stream>>>(xb, W0, W1, b0, b1,
                                            h0a, h0b, h1a, h1b, c0, c1, cnt, out);
}
